// RGCNModel_74302934220891
// MI455X (gfx1250) — compile-verified, matrix operand order checked by reading, not run on hardware
//
#include <hip/hip_runtime.h>
#include <hip/hip_bf16.h>
#include <cstddef>

// ---------------- problem constants (match reference) ----------------
#define NN   100000   // nodes
#define DD   128      // feature dim
#define RR2  100      // relations (incl. inverse)
#define NBB  4        // bases
#define EE   1000000  // edges
#define LL   2        // layers

typedef __attribute__((ext_vector_type(16))) _Float16 v16h;
typedef __attribute__((ext_vector_type(8)))  _Float16 v8h;
typedef __attribute__((ext_vector_type(8)))  float    v8f;

// Packed B layout: for a 128x128 weight matrix, fragment-linear f16:
//   packed[t][s][lane][i],  t = n-tile (0..7), s = k-step (0..3),
//   lane 0..31, i 0..15  ->  32 KB per matrix; each lane's fragment is a
//   contiguous 32 B chunk; a wave's fragment read is one coalesced 1 KB.
// Element mapping (matches v_wmma_f32_16x16x32_f16 B layout):
//   n  = t*16 + (lane & 15)
//   kb = (lane >> 4) * 8
//   k  = s*32 + kb + (i < 8 ? i : i + 8)
#define PACKED_MAT_HALVES (8 * 4 * 32 * 16)   // 16384 halves = 32 KB

template <bool TRANS>
__global__ void rgcn_pack_b(const float* __restrict__ src, _Float16* __restrict__ dst) {
  const int slot = blockIdx.x * blockDim.x + threadIdx.x;  // 0..1023
  if (slot >= 1024) return;
  const int lane = slot & 31;
  const int s    = (slot >> 5) & 3;
  const int t    = slot >> 7;
  const int n    = t * 16 + (lane & 15);
  const int kb   = (lane >> 4) * 8;
  _Float16* o = dst + (size_t)slot * 16;
#pragma unroll
  for (int i = 0; i < 16; ++i) {
    const int k = s * 32 + kb + (i < 8 ? i : i + 8);
    const float v = TRANS ? src[(size_t)n * 128 + k] : src[(size_t)k * 128 + n];
    o[i] = (_Float16)v;
  }
}

// ---------------------------------------------------------------------
// WMMA GEMM:  C[M x Nc] = A[M x 128] * B  (+ bias per column, optional)
//   AT: float (converted in-kernel) or _Float16 (direct fragment loads).
//   CT: float or _Float16 output.
//   Bp: packed fragment-linear f16; Nc/128 packed matrices concatenated.
// Per block: the 32 KB packed-B strip for this column range is staged into
// LDS once (all 8 waves share it), then each wave runs a 16x128 strip:
// one A fragment feeds 8 N-tiles, K fully unrolled -> 32 v_wmma per wave.
// ---------------------------------------------------------------------
template <typename AT, typename CT, int LDC, bool HAS_BIAS, bool ROWGUARD>
__global__ void rgcn_wmma_gemm(const AT* __restrict__ A,
                               const _Float16* __restrict__ Bp,
                               const float* __restrict__ bias,
                               CT* __restrict__ C, int M) {
  const int lane  = threadIdx.x & 31;
  const int wave  = threadIdx.x >> 5;
  const int m0    = blockIdx.y * 128 + wave * 16;  // wave's M tile
  const int nbase = blockIdx.x * 128;              // block's N strip (8 tiles)
  const int gtile0 = nbase >> 4;

  // ---- stage this block's packed-B strip (32 KB) into LDS ----
  __shared__ __align__(32) _Float16 Bsh[PACKED_MAT_HALVES];
  {
    const float4* __restrict__ srcv =
        reinterpret_cast<const float4*>(Bp + (size_t)gtile0 * (4 * 512));
    float4* dstv = reinterpret_cast<float4*>(Bsh);
    for (int i = threadIdx.x; i < PACKED_MAT_HALVES / 8; i += 256)
      dstv[i] = srcv[i];
  }
  __syncthreads();

  if (m0 >= M) return;                             // wave-uniform exit

  const int half16 = lane >> 4;                    // 0: lanes 0-15, 1: lanes 16-31
  const int l15    = lane & 15;
  const int kb     = half16 * 8;                   // per-half K sub-base

  int arow = m0 + l15;                             // A row owned by this lane
  if (ROWGUARD && arow >= M) arow = M - 1;         // clamp (stores guarded below)
  const AT* __restrict__ Arow = A + (size_t)arow * 128;

  v8f acc[8] = {};

#pragma unroll
  for (int s = 0; s < 4; ++s) {
    // ---- A fragment: K = s*32+kb+[0..7] and s*32+kb+16+[0..7]
    v16h a;
    const AT* Ap = Arow + s * 32 + kb;
    if constexpr (sizeof(AT) == 2) {
      const v8h lo = *reinterpret_cast<const v8h*>(Ap);
      const v8h hi = *reinterpret_cast<const v8h*>(Ap + 16);
#pragma unroll
      for (int i = 0; i < 8; ++i) { a[i] = lo[i]; a[i + 8] = hi[i]; }
    } else {
#pragma unroll
      for (int i = 0; i < 8; ++i) {
        a[i]     = (_Float16)Ap[i];
        a[i + 8] = (_Float16)Ap[16 + i];
      }
    }
#pragma unroll
    for (int t = 0; t < 8; ++t) {
      // fragment from LDS: contiguous 32 B per lane
      const v16h b = *reinterpret_cast<const v16h*>(
          &Bsh[(((t * 4 + s) << 9) + (lane << 4))]);
      acc[t] = __builtin_amdgcn_wmma_f32_16x16x32_f16(
          false, a, false, b, (short)0, acc[t], false, false);
    }
  }

  // ---- epilogue: C layout: VGPR r -> row rbase+r, col = tile + l15
  const int rbase = m0 + half16 * 8;
  CT* __restrict__ Crow = C + (size_t)rbase * LDC + nbase + l15;
#pragma unroll
  for (int t = 0; t < 8; ++t) {
    const float bv = HAS_BIAS ? bias[nbase + t * 16 + l15] : 0.0f;
#pragma unroll
    for (int r = 0; r < 8; ++r) {
      if (!ROWGUARD || (rbase + r) < M)
        Crow[(size_t)r * LDC + t * 16] = (CT)(acc[t][r] + bv);
    }
  }
}

// ---------------- fill u32 with zeros ----------------
__global__ void rgcn_zero_u32(unsigned* __restrict__ p, long n) {
  long i = (long)blockIdx.x * blockDim.x + threadIdx.x;
  const long stride = (long)gridDim.x * blockDim.x;
  for (; i < n; i += stride) p[i] = 0u;
}

// ---------------- count edges per (dst, relation) bucket ----------------
__global__ void rgcn_count_edges(const int* __restrict__ dst,
                                 const int* __restrict__ et,
                                 unsigned* __restrict__ cnt, long E) {
  long i = (long)blockIdx.x * blockDim.x + threadIdx.x;
  const long stride = (long)gridDim.x * blockDim.x;
  for (; i < E; i += stride) {
    atomicAdd(&cnt[(size_t)dst[i] * RR2 + et[i]], 1u);
  }
}

// ---------------------------------------------------------------------
// Edge aggregation: out[dst] += (1/cnt) * sum_b comp[et,b] * z[src, b*128+:]
// 32 lanes x float4 per edge (b128 loads of the contiguous 2KB z row),
// 8 edge slots per block. Prefetch next edge's z row (global_prefetch_b8).
// ---------------------------------------------------------------------
__global__ void rgcn_edge_agg(const float* __restrict__ z,
                              const int* __restrict__ src,
                              const int* __restrict__ dst,
                              const int* __restrict__ et,
                              const float* __restrict__ comp_l,
                              const unsigned* __restrict__ cnt,
                              float* __restrict__ out, long E) {
  const int lane = threadIdx.x;                      // 0..31 -> features lane*4..+3
  long e = (long)blockIdx.x * blockDim.y + threadIdx.y;
  const long stride = (long)gridDim.x * blockDim.y;
  for (; e < E; e += stride) {
    const int s = src[e];
    const int d = dst[e];
    const int t = et[e];

    // prefetch next iteration's z row into cache
    const long en = e + stride;
    if (en < E) {
      const int sn = src[en];
      __builtin_prefetch(z + (size_t)sn * (NBB * DD) + lane * 4, 0, 0);
    }

    const float c0 = comp_l[t * NBB + 0];
    const float c1 = comp_l[t * NBB + 1];
    const float c2 = comp_l[t * NBB + 2];
    const float c3 = comp_l[t * NBB + 3];
    const unsigned n = cnt[(size_t)d * RR2 + t];
    const float norm = 1.0f / fmaxf((float)n, 1.0f);

    const float4* zp = reinterpret_cast<const float4*>(z + (size_t)s * (NBB * DD)) + lane;
    const float4 z0 = zp[0];
    const float4 z1 = zp[32];
    const float4 z2 = zp[64];
    const float4 z3 = zp[96];

    float4 v;
    v.x = norm * (c0 * z0.x + c1 * z1.x + c2 * z2.x + c3 * z3.x);
    v.y = norm * (c0 * z0.y + c1 * z1.y + c2 * z2.y + c3 * z3.y);
    v.z = norm * (c0 * z0.z + c1 * z1.z + c2 * z2.z + c3 * z3.z);
    v.w = norm * (c0 * z0.w + c1 * z1.w + c2 * z2.w + c3 * z3.w);

    float* op = out + (size_t)d * DD + lane * 4;
    atomicAdd(op + 0, v.x);
    atomicAdd(op + 1, v.y);
    atomicAdd(op + 2, v.z);
    atomicAdd(op + 3, v.w);
  }
}

// ---- tanh f32 -> f16 (produces the next layer's f16 A operand) ----
__global__ void rgcn_tanh_h(const float* __restrict__ in,
                            _Float16* __restrict__ outh, long n) {
  long i = (long)blockIdx.x * blockDim.x + threadIdx.x;
  const long stride = (long)gridDim.x * blockDim.x;
  for (; i < n; i += stride) outh[i] = (_Float16)tanhf(in[i]);
}

// ---------------------------------------------------------------------
extern "C" void kernel_launch(void* const* d_in, const int* in_sizes, int n_in,
                              void* d_out, int out_size, void* d_ws, size_t ws_size,
                              hipStream_t stream) {
  (void)in_sizes; (void)n_in; (void)out_size; (void)ws_size;

  const float* init_embed = (const float*)d_in[0];   // [N, D]
  const float* init_rel   = (const float*)d_in[1];   // [R2, D]
  const float* w_rel      = (const float*)d_in[2];   // [D, D]
  const float* in_W       = (const float*)d_in[3];   // [D, D] (out,in)
  const float* in_b       = (const float*)d_in[4];   // [D]
  const float* basis      = (const float*)d_in[5];   // [L, NB, D, D]
  const float* comp       = (const float*)d_in[6];   // [L, R2, NB]
  const float* root       = (const float*)d_in[7];   // [L, D, D]
  const float* bias       = (const float*)d_in[8];   // [L, D]
  const int*   edge_index = (const int*)d_in[9];     // [2, E]
  const int*   edge_type  = (const int*)d_in[10];    // [E]

  const int* src = edge_index;
  const int* dst = edge_index + EE;

  float* out = (float*)d_out;                        // x [N*D] then r [R2*D]

  // ---- workspace carve-up (fully re-initialized every call) ----
  size_t off = 0;
  auto carve = [&](size_t bytes) {
    void* p = (char*)d_ws + off;
    off += (bytes + 255) & ~(size_t)255;
    return p;
  };
  _Float16* x0h = (_Float16*)carve((size_t)NN * DD * sizeof(_Float16)); // layer-0 A (f16)
  float*    x1  = (float*)carve((size_t)NN * DD * sizeof(float));       // layer-0 out (f32)
  _Float16* x1h = (_Float16*)carve((size_t)NN * DD * sizeof(_Float16)); // layer-1 A (f16)
  float*    z   = (float*)carve((size_t)NN * NBB * DD * sizeof(float)); // [N][NB*128]
  unsigned* cnt = (unsigned*)carve((size_t)NN * RR2 * sizeof(unsigned));// 10M buckets
  // packed f16 weights: in_W^T, {basis[l] (4), root[l]} x2, w_rel  -> 12 mats
  _Float16* pw_in    = (_Float16*)carve((size_t)PACKED_MAT_HALVES * sizeof(_Float16));
  _Float16* pw_bas0  = (_Float16*)carve((size_t)PACKED_MAT_HALVES * NBB * sizeof(_Float16));
  _Float16* pw_root0 = (_Float16*)carve((size_t)PACKED_MAT_HALVES * sizeof(_Float16));
  _Float16* pw_bas1  = (_Float16*)carve((size_t)PACKED_MAT_HALVES * NBB * sizeof(_Float16));
  _Float16* pw_root1 = (_Float16*)carve((size_t)PACKED_MAT_HALVES * sizeof(_Float16));
  _Float16* pw_rel   = (_Float16*)carve((size_t)PACKED_MAT_HALVES * sizeof(_Float16));
  _Float16* pw_bas[LL]   = {pw_bas0, pw_bas1};
  _Float16* pw_rootl[LL] = {pw_root0, pw_root1};

  // ---- 0) pack all weight matrices to fragment-linear f16 ----
  rgcn_pack_b<true><<<4, 256, 0, stream>>>(in_W, pw_in);   // B(k,n)=in_W[n][k]
  for (int l = 0; l < LL; ++l) {
    for (int b = 0; b < NBB; ++b)
      rgcn_pack_b<false><<<4, 256, 0, stream>>>(
          basis + ((size_t)l * NBB + b) * DD * DD,
          pw_bas[l] + (size_t)b * PACKED_MAT_HALVES);
    rgcn_pack_b<false><<<4, 256, 0, stream>>>(root + (size_t)l * DD * DD, pw_rootl[l]);
  }
  rgcn_pack_b<false><<<4, 256, 0, stream>>>(w_rel, pw_rel);

  const dim3 gblk(256);
  const int  mby = (NN + 127) / 128;   // 782

  // 1) input linear: x0h = f16( init_embed @ in_W^T + in_b )
  rgcn_wmma_gemm<float, _Float16, 128, true, false>
      <<<dim3(1, mby), gblk, 0, stream>>>(init_embed, pw_in, in_b, x0h, NN);

  // 2) per-(dst,rel) counts (shared by both layers)
  rgcn_zero_u32<<<2048, 256, 0, stream>>>(cnt, (long)NN * RR2);
  rgcn_count_edges<<<2048, 256, 0, stream>>>(dst, edge_type, cnt, (long)EE);

  // 3) layers
  const _Float16* xh = x0h;
  for (int l = 0; l < LL; ++l) {
    const float* comp_l = comp + (size_t)l * RR2 * NBB;
    const float* bias_l = bias + (size_t)l * DD;
    float* out_l = (l == LL - 1) ? out : x1;

    // z = x @ [basis_0 | basis_1 | basis_2 | basis_3]  (fused, Nc = 512)
    rgcn_wmma_gemm<_Float16, float, 512, false, false>
        <<<dim3(4, mby), gblk, 0, stream>>>(xh, pw_bas[l], nullptr, z, NN);

    // out_l = x @ root_l + bias_l  (baseline; edges atomically add on top)
    rgcn_wmma_gemm<_Float16, float, 128, true, false>
        <<<dim3(1, mby), gblk, 0, stream>>>(xh, pw_rootl[l], bias_l, out_l, NN);

    // out_l[dst] += norm * sum_b comp[et,b] * z[src]
    rgcn_edge_agg<<<8192, dim3(32, 8), 0, stream>>>(z, src, dst, edge_type,
                                                    comp_l, cnt, out_l, (long)EE);

    if (l < LL - 1) {
      rgcn_tanh_h<<<4096, 256, 0, stream>>>(out_l, x1h, (long)NN * DD);
      xh = x1h;
    }
  }

  // 4) r = init_rel @ w_rel  ->  out[N*D ...]
  rgcn_wmma_gemm<float, float, 128, false, true>
      <<<dim3(1, 1), gblk, 0, stream>>>(init_rel, pw_rel, nullptr,
                                        out + (size_t)NN * DD, RR2);
}
